// QLoRABlock_23029614641599
// MI455X (gfx1250) — compile-verified, matrix operand order checked by reading, not run on hardware
//
#include <hip/hip_runtime.h>
#include <math.h>

// ---------------------------------------------------------------------------
// QLoRA transformer block for MI455X (gfx1250, wave32, WMMA).
// NF4->f16 dequant once into workspace; all big GEMMs are double-buffered
// v_wmma_f32_16x16x32_f16 pipelines. Global->LDS uses CDNA5 async-to-LDS
// (ASYNCcnt) when the builtin exists, else a load+ds_store fallback.
// LoRA (rank 16) is folded in as one extra zero-padded K-chunk per weight set.
// ---------------------------------------------------------------------------

typedef __attribute__((ext_vector_type(16))) _Float16 v16h;
typedef __attribute__((ext_vector_type(8)))  _Float16 v8h;
typedef __attribute__((ext_vector_type(4)))  _Float16 v4h;
typedef __attribute__((ext_vector_type(8)))  float    v8f;
typedef __attribute__((ext_vector_type(4)))  int      v4i;

#define LDS_STRIDE 40   // halves; 80B row stride keeps b128 chunks 16B aligned
#define XSZ (128 * LDS_STRIDE)

#if __has_builtin(__builtin_amdgcn_global_load_async_to_lds_b128) && \
    __has_builtin(__builtin_amdgcn_s_wait_asynccnt)
#define HAVE_ASYNC_LDS 1
#else
#define HAVE_ASYNC_LDS 0
#endif

__device__ __forceinline__ void ld_g2l_b128(const _Float16* g, _Float16* l) {
#if HAVE_ASYNC_LDS
  typedef __attribute__((address_space(1))) v4i* gp_t;   // global (AS1)
  typedef __attribute__((address_space(3))) v4i* lp_t;   // LDS (AS3)
  __builtin_amdgcn_global_load_async_to_lds_b128((gp_t)g, (lp_t)l, 0, 0);
#else
  *(v8h*)l = *(const v8h*)g;
#endif
}

__device__ __forceinline__ void wait_g2l() {
#if HAVE_ASYNC_LDS
  __builtin_amdgcn_s_wait_asynccnt(0);
#endif
}

// ---------------- NF4 codebook (branchless cndmask tree) -------------------
__device__ __forceinline__ float nf4_f32(int ci) {
  unsigned c = (unsigned)ci & 15u;
  float a = (c & 1u) ? -0.6961928009986877f  : -1.0f;
  float b = (c & 1u) ? -0.39491748809814453f : -0.5250730514526367f;
  float d = (c & 1u) ? -0.18477343022823334f : -0.28444138169288635f;
  float e = (c & 1u) ?  0.0f                 : -0.09105003625154495f;
  float f = (c & 1u) ?  0.16093020141124725f :  0.07958029955625534f;
  float g = (c & 1u) ?  0.33791524171829224f :  0.24611230194568634f;
  float h = (c & 1u) ?  0.5626170039176941f  :  0.44070982933044434f;
  float i = (c & 1u) ?  1.0f                 :  0.7229568362236023f;
  float ab = (c & 2u) ? b : a;
  float de = (c & 2u) ? e : d;
  float fg = (c & 2u) ? g : f;
  float hi = (c & 2u) ? i : h;
  float lo  = (c & 4u) ? de : ab;
  float hi2 = (c & 4u) ? hi : fg;
  return (c & 8u) ? hi2 : lo;
}

// ---------------- elementwise f32 -> f16 cast ------------------------------
__global__ __launch_bounds__(256) void cast_f32_to_f16(const float* __restrict__ in,
                                                       _Float16* __restrict__ out,
                                                       long long n) {
  long long i = ((long long)blockIdx.x * blockDim.x + threadIdx.x) * 4;
  if (i >= n) return;
  float4 v = *(const float4*)(in + i);
  v4h o; o[0] = (_Float16)v.x; o[1] = (_Float16)v.y;
         o[2] = (_Float16)v.z; o[3] = (_Float16)v.w;
  *(v4h*)(out + i) = o;
}

// ---------------- NF4 dequant: codes[rows,K] + per-64 scales -> f16 --------
__global__ __launch_bounds__(256) void nf4_dequant(const int* __restrict__ codes,
                                                   const float* __restrict__ scales,
                                                   _Float16* __restrict__ out,
                                                   long long total, int K) {
  long long base = ((long long)blockIdx.x * blockDim.x + threadIdx.x) * 8;
  if (base >= total) return;
  long long row = base / K;
  int k = (int)(base - row * (long long)K);
  float s = scales[row * (long long)(K >> 6) + (k >> 6)];
  int4 c0 = *(const int4*)(codes + base);
  int4 c1 = *(const int4*)(codes + base + 4);
  v8h o;
  o[0] = (_Float16)(nf4_f32(c0.x) * s);
  o[1] = (_Float16)(nf4_f32(c0.y) * s);
  o[2] = (_Float16)(nf4_f32(c0.z) * s);
  o[3] = (_Float16)(nf4_f32(c0.w) * s);
  o[4] = (_Float16)(nf4_f32(c1.x) * s);
  o[5] = (_Float16)(nf4_f32(c1.y) * s);
  o[6] = (_Float16)(nf4_f32(c1.z) * s);
  o[7] = (_Float16)(nf4_f32(c1.w) * s);
  *(v8h*)(out + base) = o;
}

// ---------------- XA = X(f16)[M,K] @ A[set][R=16,K]^T  (tiny) --------------
__global__ __launch_bounds__(256) void lora_xa(const _Float16* __restrict__ X,
                                               const float* __restrict__ A,
                                               float* __restrict__ XA,
                                               int M, int K) {
  int j = blockIdx.y;
  int m = blockIdx.x * 16 + (threadIdx.x >> 4);
  int r = threadIdx.x & 15;
  const float* Aj = A + ((size_t)j * 16 + r) * (size_t)K;
  const _Float16* Xm = X + (size_t)m * K;
  float s = 0.f;
#pragma unroll 4
  for (int k = 0; k < K; ++k) s += (float)Xm[k] * Aj[k];
  XA[((size_t)j * M + m) * 16 + r] = s;
}

// ---------------- LDS fragment fetch (16-bit A/B layout, two b128) ---------
__device__ __forceinline__ v16h load_frag(const _Float16* s, int row0, int lane) {
  int r  = row0 + (lane & 15);
  int kb = (lane >> 4) << 3;             // lanes 16-31 start at K=8
  const _Float16* p = s + r * LDS_STRIDE + kb;
  v8h c0 = *(const v8h*)(p);             // K = kb .. kb+7
  v8h c1 = *(const v8h*)(p + 16);        // K = kb+16 .. kb+23
  return __builtin_shufflevector(c0, c1, 0,1,2,3,4,5,6,7,8,9,10,11,12,13,14,15);
}

// ---------------- LoRA chunk fill: cols 0-15 = f32 data, 16-31 = 0 ---------
__device__ __forceinline__ void fill_lora_rows(_Float16* dst, const float* src,
                                               int base, int idx) {
  int r = idx >> 1, hf = idx & 1;
  if (hf == 0) {
    const float* p = src + (size_t)(base + r) * 16;
    v8h a, b;
#pragma unroll
    for (int e = 0; e < 8; ++e) { a[e] = (_Float16)p[e]; b[e] = (_Float16)p[8 + e]; }
    *(v8h*)(dst + r * LDS_STRIDE)     = a;
    *(v8h*)(dst + r * LDS_STRIDE + 8) = b;
  } else {
    v8h z = {};
    *(v8h*)(dst + r * LDS_STRIDE + 16) = z;
    *(v8h*)(dst + r * LDS_STRIDE + 24) = z;
  }
}

// ---------------- main GEMM: out = sum_j (X @ W16_j^T + XA_j @ B_j^T) ------
__global__ __launch_bounds__(256) void gemm_wmma_lora(
    const _Float16* __restrict__ X,    // [M,K] f16
    const _Float16* __restrict__ W,    // [nsets,N,K] f16 (pre-dequantized)
    const float*    __restrict__ XA,   // [nsets,M,16]
    const float*    __restrict__ BL,   // [nsets,N,16]
    const float*    __restrict__ resid,// [M,N] f32 or null
    float*          __restrict__ outF, // [M,N] f32 or null
    _Float16*       __restrict__ outH, // [M,N] f16 or null
    int M, int N, int K, int nsets, float oscale) {
  __shared__ alignas(16) _Float16 sX[2 * XSZ];   // double-buffered
  __shared__ alignas(16) _Float16 sW[2 * XSZ];

  const int m0 = blockIdx.x * 128, n0 = blockIdx.y * 128;
  const int tid = threadIdx.x, lane = tid & 31, wv = tid >> 5;
  const int wm = wv & 3, wn = wv >> 2;           // wave tile: 32(M) x 64(N)
  const int kchunks = K >> 5;
  const int cps = kchunks + 1;                   // chunks per set (+1 = LoRA)
  const int total = nsets * cps;

  v8f acc[2][4] = {};

  // fill chunk c into buffer b
  auto fill = [&](int c, int b) {
    _Float16* dX = sX + b * XSZ;
    _Float16* dW = sW + b * XSZ;
    int j = c / cps, kk = c - j * cps;
    if (kk < kchunks) {
      const _Float16* Wj = W + (size_t)j * N * K;
      int k0 = kk << 5;
#pragma unroll
      for (int i = tid; i < 512; i += 256) {     // 128 rows x 4 b128 chunks
        int r = i >> 2, ch = (i & 3) << 3;
        ld_g2l_b128(X  + (size_t)(m0 + r) * K + k0 + ch, dX + r * LDS_STRIDE + ch);
        ld_g2l_b128(Wj + (size_t)(n0 + r) * K + k0 + ch, dW + r * LDS_STRIDE + ch);
      }
    } else {                                     // rank-16 LoRA chunk
      fill_lora_rows(dX, XA + (size_t)j * M * 16, m0, tid);
      fill_lora_rows(dW, BL + (size_t)j * N * 16, n0, tid);
    }
  };

  fill(0, 0);
  for (int c = 0; c < total; ++c) {
    const int b = c & 1;
    wait_g2l();
    __syncthreads();
    if (c + 1 < total) fill(c + 1, b ^ 1);

    const _Float16* cX = sX + b * XSZ;
    const _Float16* cW = sW + b * XSZ;
    v16h a[2], bb[4];
#pragma unroll
    for (int mi = 0; mi < 2; ++mi) a[mi] = load_frag(cX, wm * 32 + mi * 16, lane);
#pragma unroll
    for (int ni = 0; ni < 4; ++ni) bb[ni] = load_frag(cW, wn * 64 + ni * 16, lane);
#pragma unroll
    for (int mi = 0; mi < 2; ++mi)
#pragma unroll
      for (int ni = 0; ni < 4; ++ni)
        acc[mi][ni] = __builtin_amdgcn_wmma_f32_16x16x32_f16(
            false, a[mi], false, bb[ni], (short)0, acc[mi][ni], false, false);
  }

  // epilogue: C layout -> lanes 0-15 rows M=r, lanes 16-31 rows M=8+r
  const int ecol = lane & 15, erow = (lane >> 4) * 8;
#pragma unroll
  for (int mi = 0; mi < 2; ++mi)
#pragma unroll
    for (int ni = 0; ni < 4; ++ni) {
      int mb = m0 + wm * 32 + mi * 16 + erow;
      int n  = n0 + wn * 64 + ni * 16 + ecol;
#pragma unroll
      for (int r = 0; r < 8; ++r) {
        size_t off = (size_t)(mb + r) * N + n;
        float v = acc[mi][ni][r] * oscale;
        if (resid) v += resid[off];
        if (outF)  outF[off] = v;
        if (outH)  outH[off] = (_Float16)v;
      }
    }
}

// ---------------- fused gate/up: h = silu(g) * u  (f16 out) ----------------
__global__ __launch_bounds__(256) void gemm_gateup(
    const _Float16* __restrict__ X,   // [M,K] f16 (x1)
    const _Float16* __restrict__ Wg, const _Float16* __restrict__ Wu, // [N,K]
    const float* __restrict__ XAg, const float* __restrict__ XAu,     // [M,16]
    const float* __restrict__ BLg, const float* __restrict__ BLu,     // [N,16]
    _Float16* __restrict__ outH,      // [M,N] f16
    int M, int N, int K) {
  __shared__ alignas(16) _Float16 sX[2 * XSZ];            // 128 rows
  __shared__ alignas(16) _Float16 sG[2 * 64 * LDS_STRIDE];// 64 rows
  __shared__ alignas(16) _Float16 sU[2 * 64 * LDS_STRIDE];

  const int m0 = blockIdx.x * 128, n0 = blockIdx.y * 64;
  const int tid = threadIdx.x, lane = tid & 31, wv = tid >> 5;
  const int wm = wv & 3, wn = wv >> 2;          // wave tile: 32(M) x 32(N)
  const int kchunks = K >> 5;
  const int total = kchunks + 2;                // + LoRA-g + LoRA-u chunks

  v8f ag[2][2] = {}, au[2][2] = {};

  auto fill = [&](int c, int b) {
    _Float16* dX = sX + b * XSZ;
    _Float16* dG = sG + b * 64 * LDS_STRIDE;
    _Float16* dU = sU + b * 64 * LDS_STRIDE;
    if (c < kchunks) {
      int k0 = c << 5;
#pragma unroll
      for (int i = tid; i < 512; i += 256) {
        int r = i >> 2, ch = (i & 3) << 3;
        ld_g2l_b128(X + (size_t)(m0 + r) * K + k0 + ch, dX + r * LDS_STRIDE + ch);
      }
      {
        int r = tid >> 2, ch = (tid & 3) << 3; // 64 rows x 4 chunks = 256
        ld_g2l_b128(Wg + (size_t)(n0 + r) * K + k0 + ch, dG + r * LDS_STRIDE + ch);
        ld_g2l_b128(Wu + (size_t)(n0 + r) * K + k0 + ch, dU + r * LDS_STRIDE + ch);
      }
    } else {
      const bool isLg = (c == kchunks);
      fill_lora_rows(dX, isLg ? XAg : XAu, m0, tid);
      if (tid < 128) fill_lora_rows(isLg ? dG : dU, isLg ? BLg : BLu, n0, tid);
    }
  };

  fill(0, 0);
  for (int c = 0; c < total; ++c) {
    const int b = c & 1;
    const bool isLg = (c == kchunks), isLu = (c == kchunks + 1);
    wait_g2l();
    __syncthreads();
    if (c + 1 < total) fill(c + 1, b ^ 1);

    const _Float16* cX = sX + b * XSZ;
    const _Float16* cG = sG + b * 64 * LDS_STRIDE;
    const _Float16* cU = sU + b * 64 * LDS_STRIDE;
    v16h a[2];
#pragma unroll
    for (int mi = 0; mi < 2; ++mi) a[mi] = load_frag(cX, wm * 32 + mi * 16, lane);
    if (!isLu) {
#pragma unroll
      for (int ni = 0; ni < 2; ++ni) {
        v16h bg = load_frag(cG, wn * 32 + ni * 16, lane);
#pragma unroll
        for (int mi = 0; mi < 2; ++mi)
          ag[mi][ni] = __builtin_amdgcn_wmma_f32_16x16x32_f16(
              false, a[mi], false, bg, (short)0, ag[mi][ni], false, false);
      }
    }
    if (!isLg) {
#pragma unroll
      for (int ni = 0; ni < 2; ++ni) {
        v16h bu = load_frag(cU, wn * 32 + ni * 16, lane);
#pragma unroll
        for (int mi = 0; mi < 2; ++mi)
          au[mi][ni] = __builtin_amdgcn_wmma_f32_16x16x32_f16(
              false, a[mi], false, bu, (short)0, au[mi][ni], false, false);
      }
    }
  }

  const int ecol = lane & 15, erow = (lane >> 4) * 8;
#pragma unroll
  for (int mi = 0; mi < 2; ++mi)
#pragma unroll
    for (int ni = 0; ni < 2; ++ni) {
      int mb = m0 + wm * 32 + mi * 16 + erow;
      int n  = n0 + wn * 32 + ni * 16 + ecol;
#pragma unroll
      for (int r = 0; r < 8; ++r) {
        float g = ag[mi][ni][r], u = au[mi][ni][r];
        float h = (g / (1.f + expf(-g))) * u;        // silu(g)*u
        outH[(size_t)(mb + r) * N + n] = (_Float16)h;
      }
    }
}

// ---------------------------------------------------------------------------
extern "C" void kernel_launch(void* const* d_in, const int* in_sizes, int n_in,
                              void* d_out, int out_size, void* d_ws, size_t ws_size,
                              hipStream_t stream) {
  (void)in_sizes; (void)n_in; (void)out_size; (void)ws_size;
  const int M = 4096, H = 4096, I = 14336, R = 16;

  const float* x           = (const float*)d_in[0];
  const int*   codes_attn  = (const int*)  d_in[1];
  const float* scales_attn = (const float*)d_in[2];
  const int*   codes_gu    = (const int*)  d_in[3];
  const float* scales_gu   = (const float*)d_in[4];
  const int*   codes_down  = (const int*)  d_in[5];
  const float* scales_down = (const float*)d_in[6];
  const float* lora_A_h    = (const float*)d_in[7];
  const float* lora_A_down = (const float*)d_in[8];
  const float* lora_B_attn = (const float*)d_in[9];
  const float* lora_B_gu   = (const float*)d_in[10];
  const float* lora_B_down = (const float*)d_in[11];
  float* out = (float*)d_out;

  // workspace carve-up (~510 MB total)
  char* ws = (char*)d_ws;
  size_t off = 0;
  _Float16* X16  = (_Float16*)(ws + off); off += (size_t)M * H * 2;   // 32MB
  _Float16* ATT16= (_Float16*)(ws + off); off += (size_t)M * H * 2;   // 32MB
  _Float16* X116 = (_Float16*)(ws + off); off += (size_t)M * H * 2;   // 32MB
  float*    X1   = (float*)   (ws + off); off += (size_t)M * H * 4;   // 64MB
  _Float16* H16  = (_Float16*)(ws + off); off += (size_t)M * I * 2;   // 112MB
  float*    XAb  = (float*)   (ws + off); off += (size_t)3 * M * 16 * 4;
  _Float16* W16  = (_Float16*)(ws + off);                              // up to 224MB

  dim3 blk(256);

  // 1) x -> f16
  cast_f32_to_f16<<<((size_t)M * H) / 4 / 256, blk, 0, stream>>>(x, X16, (long long)M * H);

  // 2) q,k,v fused: dequant 3 sets, XA, GEMM (sum/3) -> ATT16
  nf4_dequant<<<(unsigned)((size_t)3 * H * H / 2048), blk, 0, stream>>>(
      codes_attn, scales_attn, W16, (long long)3 * H * H, H);
  lora_xa<<<dim3(M / 16, 3), blk, 0, stream>>>(X16, lora_A_h, XAb, M, H);
  gemm_wmma_lora<<<dim3(M / 128, H / 128), blk, 0, stream>>>(
      X16, W16, XAb, lora_B_attn, nullptr, nullptr, ATT16,
      M, H, H, 3, 1.f / 3.f);

  // 3) o-proj + residual: x1 = x + o  (f32 + f16 copies)
  nf4_dequant<<<(unsigned)((size_t)H * H / 2048), blk, 0, stream>>>(
      codes_attn + (size_t)3 * H * H, scales_attn + (size_t)3 * H * (H / 64),
      W16, (long long)H * H, H);
  lora_xa<<<dim3(M / 16, 1), blk, 0, stream>>>(ATT16, lora_A_h + (size_t)3 * R * H, XAb, M, H);
  gemm_wmma_lora<<<dim3(M / 128, H / 128), blk, 0, stream>>>(
      ATT16, W16, XAb, lora_B_attn + (size_t)3 * H * R, x, X1, X116,
      M, H, H, 1, 1.f);

  // 4) fused gate/up -> h = silu(g)*u (f16)
  nf4_dequant<<<(unsigned)((size_t)2 * I * H / 2048), blk, 0, stream>>>(
      codes_gu, scales_gu, W16, (long long)2 * I * H, H);
  lora_xa<<<dim3(M / 16, 2), blk, 0, stream>>>(X116, lora_A_h + (size_t)4 * R * H, XAb, M, H);
  gemm_gateup<<<dim3(M / 128, I / 64), blk, 0, stream>>>(
      X116, W16, W16 + (size_t)I * H,
      XAb, XAb + (size_t)M * 16,
      lora_B_gu, lora_B_gu + (size_t)I * R,
      H16, M, I, H);

  // 5) down-proj + residual: out = x1 + d
  nf4_dequant<<<(unsigned)((size_t)H * I / 2048), blk, 0, stream>>>(
      codes_down, scales_down, W16, (long long)H * I, I);
  lora_xa<<<dim3(M / 16, 1), blk, 0, stream>>>(H16, lora_A_down, XAb, M, I);
  gemm_wmma_lora<<<dim3(M / 128, H / 128), blk, 0, stream>>>(
      H16, W16, XAb, lora_B_down, X1, out, nullptr,
      M, H, I, 1, 1.f);
}
